// QSAR_8718783611149
// MI455X (gfx1250) — matrix operand order repaired and checked by reading, not compiled, not run on hardware
//
#include <hip/hip_runtime.h>
#include <math.h>

typedef _Float16 half8 __attribute__((ext_vector_type(8)));
typedef _Float16 v16h  __attribute__((ext_vector_type(16)));
typedef float    v8f   __attribute__((ext_vector_type(8)));

#define NB 2048
#define NATOM 60
#define DEG 6
#define HDIM 128

// ---------------------------------------------------------------------------
// Weight prep: (Dn, KIN, 128) f32 -> (Dn, 128, KPAD) f16, K zero-padded.
// ---------------------------------------------------------------------------
__global__ void prep_w_kernel(const float* __restrict__ src, _Float16* __restrict__ dst,
                              int Dn, int KIN, int KPAD) {
    int total = Dn * HDIM * KPAD;
    for (int idx = blockIdx.x * blockDim.x + threadIdx.x; idx < total;
         idx += gridDim.x * blockDim.x) {
        int d = idx / (HDIM * KPAD);
        int r = idx - d * (HDIM * KPAD);
        int o = r / KPAD;
        int k = r - o * KPAD;
        float v = (k < KIN) ? src[((size_t)d * KIN + k) * HDIM + o] : 0.0f;
        dst[idx] = (_Float16)v;
    }
}

// ---------------------------------------------------------------------------
// bondsum[b,n,f] = sum_j bonds[b,n,j,f]   (B,N,6,6) -> (B,N,6)
// ---------------------------------------------------------------------------
__global__ void bondsum_kernel(const float* __restrict__ bonds, float* __restrict__ bs) {
    int idx = blockIdx.x * blockDim.x + threadIdx.x;
    int total = NB * NATOM * DEG;
    if (idx >= total) return;
    int bn = idx / DEG;
    int f  = idx - bn * DEG;
    const float* p = bonds + (size_t)bn * 36 + f;
    float s = 0.0f;
#pragma unroll
    for (int j = 0; j < DEG; ++j) s += p[j * DEG];
    bs[idx] = s;
}

// ---------------------------------------------------------------------------
// Degree-selective graph conv (one block = one graph, 256 thr = 8 waves).
// feat = [self+neighbor sum of xin (INF)] ++ bondsum(6), zero-padded to KPAD.
// out[b,n,:] = relu(feat @ W[deg[n]] + bias[deg[n]])
// ---------------------------------------------------------------------------
template <int INF, int KPAD, int KT>
__global__ void conv_kernel(const float* __restrict__ xin,
                            const float* __restrict__ bs,
                            const int*   __restrict__ edges,
                            const _Float16* __restrict__ wT,   // (6,128,KPAD) f16
                            const float* __restrict__ bias,    // (6,128)
                            float* __restrict__ xout) {        // (B,60,128)
    __shared__ __align__(16) _Float16 featLDS[64 * KPAD];
    __shared__ int edgeLDS[NATOM * DEG];
    __shared__ int degLDS[64];

    const int b   = blockIdx.x;
    const int tid = threadIdx.x;

    // stage edges
    for (int i = tid; i < NATOM * DEG; i += 256)
        edgeLDS[i] = edges[(size_t)b * NATOM * DEG + i];
    __syncthreads();

    if (tid < 64) {
        int dg = 0;
        if (tid < NATOM) {
#pragma unroll
            for (int j = 0; j < DEG; ++j) dg += (edgeLDS[tid * DEG + j] != -1);
        } else {
            dg = -1;
        }
        degLDS[tid] = dg;
    }
    __syncthreads();

    // build feature tile (64 x KPAD) in f16
    for (int idx = tid; idx < 64 * KPAD; idx += 256) {
        int n = idx / KPAD;
        int k = idx - n * KPAD;
        float v = 0.0f;
        if (n < NATOM) {
            if (k < INF) {
                v = xin[((size_t)b * NATOM + n) * INF + k];
#pragma unroll
                for (int j = 0; j < DEG; ++j) {
                    int e = edgeLDS[n * DEG + j];
                    if (e >= 0) v += xin[((size_t)b * NATOM + e) * INF + k];
                }
            } else if (k < INF + DEG) {
                v = bs[((size_t)b * NATOM + n) * DEG + (k - INF)];
            }
        }
        featLDS[idx] = (_Float16)v;
    }
    __syncthreads();

    const int lane = tid & 31;
    const int wave = tid >> 5;          // 0..7 -> N tile
    const int cn   = lane & 15;
    const int hi   = lane >> 4;         // 0/1 half of wave
    const int col  = wave * 16 + cn;

    for (int d = 0; d < DEG; ++d) {
        v8f acc[4];
#pragma unroll
        for (int mt = 0; mt < 4; ++mt) acc[mt] = (v8f)(0.0f);

        for (int kt = 0; kt < KT; ++kt) {
            // B fragment: 16 contiguous halves at (d,col, kt*32 + hi*16)
            const _Float16* wp =
                wT + ((size_t)(d * HDIM + col) * KPAD) + kt * 32 + hi * 16;
            v16h bf = *(const v16h*)wp;
#pragma unroll
            for (int mt = 0; mt < 4; ++mt) {
                const _Float16* ap = &featLDS[(mt * 16 + cn) * KPAD + kt * 32 + hi * 16];
                half8 a0 = *(const half8*)(ap);
                half8 a1 = *(const half8*)(ap + 8);
                v16h af = __builtin_shufflevector(a0, a1, 0, 1, 2, 3, 4, 5, 6, 7,
                                                          8, 9, 10, 11, 12, 13, 14, 15);
                acc[mt] = __builtin_amdgcn_wmma_f32_16x16x32_f16(
                    false, af, false, bf, (short)0, acc[mt], false, false);
            }
        }

        const float bv = bias[d * HDIM + col];
#pragma unroll
        for (int mt = 0; mt < 4; ++mt) {
#pragma unroll
            for (int v = 0; v < 8; ++v) {
                int row = mt * 16 + v + hi * 8;
                if (row < NATOM && degLDS[row] == d) {
                    float val = acc[mt][v] + bv;
                    xout[((size_t)b * NATOM + row) * HDIM + col] = val > 0.0f ? val : 0.0f;
                }
            }
        }
    }
}

// ---------------------------------------------------------------------------
// Pool: max over self + valid neighbors, zero when deg==0.
// ---------------------------------------------------------------------------
__global__ void pool_kernel(const float* __restrict__ xin,
                            const int* __restrict__ edges,
                            float* __restrict__ xout) {
    int idx = blockIdx.x * blockDim.x + threadIdx.x;
    int total = NB * NATOM * HDIM;
    if (idx >= total) return;
    int bn = idx / HDIM;
    int f  = idx - bn * HDIM;
    const int* e = edges + (size_t)bn * DEG;
    int bbase = (bn / NATOM) * NATOM;
    float m = xin[(size_t)bn * HDIM + f];
    int dg = 0;
#pragma unroll
    for (int j = 0; j < DEG; ++j) {
        int ej = e[j];
        if (ej >= 0) {
            ++dg;
            float t = xin[((size_t)bbase + ej) * HDIM + f];
            m = fmaxf(m, t);
        }
    }
    xout[idx] = (dg > 0) ? m : 0.0f;
}

// ---------------------------------------------------------------------------
// Graph output: fp = sum_n mask_n * tanh([x_n ++ bondsum_n] @ Wo + bo)
// out = sigmoid(fp @ Wfc + bfc)
// ---------------------------------------------------------------------------
__global__ void gout_kernel(const float* __restrict__ xin,      // (B,60,128)
                            const float* __restrict__ bs,       // (B,60,6)
                            const int*   __restrict__ edges,
                            const _Float16* __restrict__ woT,   // (128,160) f16
                            const float* __restrict__ bo,       // (128)
                            const float* __restrict__ Wfc,      // (128,12)
                            const float* __restrict__ bfc,      // (12)
                            float* __restrict__ out) {          // (B,12)
    const int KPAD = 160, KT = 5, NCLS = 12;
    __shared__ __align__(16) _Float16 featLDS[64 * 160];
    __shared__ int degLDS[64];
    __shared__ float colsumLDS[2 * HDIM];
    __shared__ float fpLDS[HDIM];

    const int b   = blockIdx.x;
    const int tid = threadIdx.x;

    if (tid < 64) {
        int dg = 0;
        if (tid < NATOM) {
            const int* e = edges + ((size_t)b * NATOM + tid) * DEG;
#pragma unroll
            for (int j = 0; j < DEG; ++j) dg += (e[j] != -1);
        }
        degLDS[tid] = dg;
    }

    for (int idx = tid; idx < 64 * KPAD; idx += 256) {
        int n = idx / KPAD;
        int k = idx - n * KPAD;
        float v = 0.0f;
        if (n < NATOM) {
            if (k < HDIM)              v = xin[((size_t)b * NATOM + n) * HDIM + k];
            else if (k < HDIM + DEG)   v = bs[((size_t)b * NATOM + n) * DEG + (k - HDIM)];
        }
        featLDS[idx] = (_Float16)v;
    }
    __syncthreads();

    const int lane = tid & 31;
    const int wave = tid >> 5;
    const int cn   = lane & 15;
    const int hi   = lane >> 4;
    const int col  = wave * 16 + cn;

    v8f acc[4];
#pragma unroll
    for (int mt = 0; mt < 4; ++mt) acc[mt] = (v8f)(0.0f);

    for (int kt = 0; kt < KT; ++kt) {
        const _Float16* wp = woT + (size_t)col * KPAD + kt * 32 + hi * 16;
        v16h bf = *(const v16h*)wp;
#pragma unroll
        for (int mt = 0; mt < 4; ++mt) {
            const _Float16* ap = &featLDS[(mt * 16 + cn) * KPAD + kt * 32 + hi * 16];
            half8 a0 = *(const half8*)(ap);
            half8 a1 = *(const half8*)(ap + 8);
            v16h af = __builtin_shufflevector(a0, a1, 0, 1, 2, 3, 4, 5, 6, 7,
                                                      8, 9, 10, 11, 12, 13, 14, 15);
            acc[mt] = __builtin_amdgcn_wmma_f32_16x16x32_f16(
                false, af, false, bf, (short)0, acc[mt], false, false);
        }
    }

    const float bv = bo[col];
    float partial = 0.0f;
#pragma unroll
    for (int mt = 0; mt < 4; ++mt) {
#pragma unroll
        for (int v = 0; v < 8; ++v) {
            int row = mt * 16 + v + hi * 8;
            if (row < NATOM && degLDS[row] > 0)
                partial += tanhf(acc[mt][v] + bv);
        }
    }
    colsumLDS[hi * HDIM + col] = partial;
    __syncthreads();

    if (tid < HDIM) fpLDS[tid] = colsumLDS[tid] + colsumLDS[HDIM + tid];
    __syncthreads();

    if (tid < NCLS) {
        float s = bfc[tid];
        for (int k = 0; k < HDIM; ++k) s += fpLDS[k] * Wfc[k * NCLS + tid];
        out[(size_t)b * NCLS + tid] = 1.0f / (1.0f + expf(-s));
    }
}

// ---------------------------------------------------------------------------
extern "C" void kernel_launch(void* const* d_in, const int* in_sizes, int n_in,
                              void* d_out, int out_size, void* d_ws, size_t ws_size,
                              hipStream_t stream) {
    const float* atoms = (const float*)d_in[0];
    const float* bonds = (const float*)d_in[1];
    const int*   edges = (const int*)d_in[2];
    const float* W1    = (const float*)d_in[3];
    const float* b1    = (const float*)d_in[4];
    const float* W2    = (const float*)d_in[5];
    const float* b2    = (const float*)d_in[6];
    const float* Wo    = (const float*)d_in[7];
    const float* bo    = (const float*)d_in[8];
    const float* Wfc   = (const float*)d_in[9];
    const float* bfc   = (const float*)d_in[10];
    float* out = (float*)d_out;

    char* ws = (char*)d_ws;
    const size_t SZ_BS  = (size_t)NB * NATOM * DEG * sizeof(float);     // 2.95 MB
    const size_t SZ_X   = (size_t)NB * NATOM * HDIM * sizeof(float);    // 62.9 MB
    const size_t SZ_W1T = (size_t)DEG * HDIM * 64 * sizeof(_Float16);
    const size_t SZ_W2T = (size_t)DEG * HDIM * 160 * sizeof(_Float16);

    float*    bs  = (float*)(ws);
    float*    X   = (float*)(ws + SZ_BS);
    float*    P   = (float*)(ws + SZ_BS + SZ_X);
    _Float16* w1t = (_Float16*)(ws + SZ_BS + 2 * SZ_X);
    _Float16* w2t = (_Float16*)(ws + SZ_BS + 2 * SZ_X + SZ_W1T);
    _Float16* wot = (_Float16*)(ws + SZ_BS + 2 * SZ_X + SZ_W1T + SZ_W2T);

    // weight prep (f32 -> padded/transposed f16)
    prep_w_kernel<<<(DEG * HDIM * 64 + 255) / 256, 256, 0, stream>>>(W1, w1t, DEG, 43, 64);
    prep_w_kernel<<<(DEG * HDIM * 160 + 255) / 256, 256, 0, stream>>>(W2, w2t, DEG, 134, 160);
    prep_w_kernel<<<(HDIM * 160 + 255) / 256, 256, 0, stream>>>(Wo, wot, 1, 134, 160);

    // bond sums
    bondsum_kernel<<<(NB * NATOM * DEG + 255) / 256, 256, 0, stream>>>(bonds, bs);

    // gcn1 -> X, pool -> P
    conv_kernel<37, 64, 2><<<NB, 256, 0, stream>>>(atoms, bs, edges, w1t, b1, X);
    pool_kernel<<<(NB * NATOM * HDIM + 255) / 256, 256, 0, stream>>>(X, edges, P);

    // gcn2 -> X, pool -> P
    conv_kernel<128, 160, 5><<<NB, 256, 0, stream>>>(P, bs, edges, w2t, b2, X);
    pool_kernel<<<(NB * NATOM * HDIM + 255) / 256, 256, 0, stream>>>(X, edges, P);

    // graph output + fc + sigmoid
    gout_kernel<<<NB, 256, 0, stream>>>(P, bs, edges, wot, bo, Wfc, bfc, out);
}
